// RGRL_11158325035024
// MI455X (gfx1250) — matrix operand order repaired and checked by reading, not compiled, not run on hardware
//
#include <hip/hip_runtime.h>
#include <hip/hip_bf16.h>

// ---------------------------------------------------------------------------
// RGRL forward for MI455X (gfx1250, wave32, WMMA).
// GEMMs in bf16 via v_wmma_f32_16x16x32_bf16; segment-sum via f32 global
// atomics; fused epilogues (bias / PReLU / 1/T scale / bf16 requant).
// ---------------------------------------------------------------------------

#define N_NODES 50000
#define E_EDGES 800000
#define D_IN    512
#define H_HID   512
#define R_REP   256
#define K_DIFF  8
#define TOPK    256
#define PH_DIM  512

typedef unsigned short u16;
typedef __bf16 bf16x16 __attribute__((ext_vector_type(16)));
typedef float  f32x8   __attribute__((ext_vector_type(8)));

union Frag { bf16x16 v; uint4 q[2]; u16 u[16]; };
union H8   { uint4 q;   u16 u[8]; };

__device__ __forceinline__ u16 f2bf(float f) {
    unsigned u = __float_as_uint(f);
    unsigned r = u + 0x7FFFu + ((u >> 16) & 1u);   // round-to-nearest-even
    return (u16)(r >> 16);
}
__device__ __forceinline__ float bf2f(u16 h) {
    return __uint_as_float(((unsigned)h) << 16);
}

// ---------------------------------------------------------------------------
// Generic bf16 GEMM: C(MxNo) = A(MxK) @ B(KxNo), row-major bf16 in/out f32.
// Epilogue: y = prelu(C + bias, *alpha) * scale ; optional f32 and bf16 out.
// Block: 256 thr = 8 waves; block tile 128x64; wave tile 32x32 (2x2 WMMA).
// Requires K%32==0, No%64==0 (true for 512/256 here).
// ---------------------------------------------------------------------------
__global__ __launch_bounds__(256) void gemm_bf16_wmma(
    const u16* __restrict__ A, const u16* __restrict__ B,
    const float* __restrict__ bias, const float* __restrict__ alpha,
    float scale, float* __restrict__ outF, u16* __restrict__ outB,
    int M, int K, int No)
{
    __shared__ u16 sA[128 * 40];   // 128 rows x 32 K-halves, pitch 40
    __shared__ u16 sB[64 * 40];    // 64 cols  x 32 K-halves (B transposed)

    const int tid  = threadIdx.x;
    const int lane = tid & 31;
    const int wid  = tid >> 5;
    const int wm   = wid & 3;      // 4 wave rows of 32
    const int wn   = wid >> 2;     // 2 wave cols of 32
    const int half = lane >> 4;    // K-half selector (WMMA layout)
    const int lr   = lane & 15;
    const int bm   = blockIdx.x * 128;
    const int bn   = blockIdx.y * 64;

    f32x8 acc[2][2] = {};

    for (int k0 = 0; k0 < K; k0 += 32) {
        // ---- stage A tile: 128x32 halves, 16B chunks ----
        for (int it = 0; it < 2; ++it) {
            int chunk = it * 256 + tid;            // 512 chunks of 8 halves
            int row = chunk >> 2, coff = (chunk & 3) * 8;
            uint4 v = {0u, 0u, 0u, 0u};
            int gr = bm + row;
            if (gr < M) v = *(const uint4*)(A + (long)gr * K + k0 + coff);
            *(uint4*)(&sA[row * 40 + coff]) = v;
        }
        // ---- stage B tile transposed: sB[n][k] ----
        {
            int kr = tid >> 3, coff = (tid & 7) * 8;
            H8 t; t.q = *(const uint4*)(B + (long)(k0 + kr) * No + bn + coff);
            #pragma unroll
            for (int j = 0; j < 8; ++j) sB[(coff + j) * 40 + kr] = t.u[j];
        }
        if (k0 + 32 < K)   // speculative prefetch of next A stripe
            __builtin_prefetch(A + (long)min(bm + (tid >> 1), M - 1) * K + k0 + 32, 0, 1);
        __syncthreads();

        // ---- fragments (layouts per CDNA5 ISA 7.12.2) ----
        Frag a[2], b[2];
        #pragma unroll
        for (int mi = 0; mi < 2; ++mi) {
            const u16* rp = &sA[(wm * 32 + mi * 16 + lr) * 40];
            a[mi].q[0] = *(const uint4*)(rp + half * 8);        // K 0..7 / 8..15
            a[mi].q[1] = *(const uint4*)(rp + 16 + half * 8);   // K 16..23 / 24..31
        }
        #pragma unroll
        for (int ni = 0; ni < 2; ++ni) {
            const u16* rp = &sB[(wn * 32 + ni * 16 + lr) * 40];
            b[ni].q[0] = *(const uint4*)(rp + half * 8);
            b[ni].q[1] = *(const uint4*)(rp + 16 + half * 8);
        }
        #pragma unroll
        for (int mi = 0; mi < 2; ++mi)
            #pragma unroll
            for (int ni = 0; ni < 2; ++ni)
                acc[mi][ni] = __builtin_amdgcn_wmma_f32_16x16x32_bf16(
                    false, a[mi].v, false, b[ni].v,
                    (short)0, acc[mi][ni], false, false);
        __syncthreads();
    }

    // ---- epilogue ----
    const float av = alpha ? alpha[0] : 0.f;
    #pragma unroll
    for (int mi = 0; mi < 2; ++mi) {
        #pragma unroll
        for (int ni = 0; ni < 2; ++ni) {
            int gc  = bn + wn * 32 + ni * 16 + lr;
            float bi = bias ? bias[gc] : 0.f;
            int gr0 = bm + wm * 32 + mi * 16 + half * 8;
            #pragma unroll
            for (int r = 0; r < 8; ++r) {
                int gr = gr0 + r;
                if (gr >= M) continue;
                float y = acc[mi][ni][r] + bi;
                if (alpha) y = (y >= 0.f) ? y : av * y;
                y *= scale;
                long o = (long)gr * No + gc;
                if (outF) outF[o] = y;
                if (outB) outB[o] = f2bf(y);
            }
        }
    }
}

// ---------------------------------------------------------------------------
// Elementwise: x = prelu(in, *alpha) (alpha optional); write f32 / bf16.
// ---------------------------------------------------------------------------
__global__ void ew_kernel(const float* __restrict__ in,
                          const float* __restrict__ alpha,
                          float* __restrict__ outF, u16* __restrict__ outB,
                          long n)
{
    long i = (long)blockIdx.x * blockDim.x + threadIdx.x;
    if (i >= n) return;
    float x = in[i];
    if (alpha) { float a = alpha[0]; x = (x >= 0.f) ? x : a * x; }
    if (outF) outF[i] = x;
    if (outB) outB[i] = f2bf(x);
}

// ---------------------------------------------------------------------------
// Edge aggregation: out[dst] += w[e] * h[src] (segment_sum). f32 atomics.
// One thread per (edge, 4 features).
// ---------------------------------------------------------------------------
__global__ void edge_agg_kernel(const int* __restrict__ ei,
                                const float* __restrict__ ew,
                                const float* __restrict__ h,
                                float* __restrict__ out, int E, int F)
{
    long idx = (long)blockIdx.x * blockDim.x + threadIdx.x;
    int per = F >> 2;
    long tot = (long)E * per;
    if (idx >= tot) return;
    int e = (int)(idx / per);
    int f = (int)(idx % per) << 2;
    int src = ei[e], dst = ei[E + e];
    float w = ew[e];
    const float4 hv = *(const float4*)(h + (long)src * F + f);
    float* o = out + (long)dst * F + f;
    __hip_atomic_fetch_add(o + 0, w * hv.x, __ATOMIC_RELAXED, __HIP_MEMORY_SCOPE_AGENT);
    __hip_atomic_fetch_add(o + 1, w * hv.y, __ATOMIC_RELAXED, __HIP_MEMORY_SCOPE_AGENT);
    __hip_atomic_fetch_add(o + 2, w * hv.z, __ATOMIC_RELAXED, __HIP_MEMORY_SCOPE_AGENT);
    __hip_atomic_fetch_add(o + 3, w * hv.w, __ATOMIC_RELAXED, __HIP_MEMORY_SCOPE_AGENT);
}

// ---------------------------------------------------------------------------
// Fused (optional PReLU) -> L2-normalize rows of (rows x 256) -> bf16.
// One wave32 per row; 8 elems/lane; shfl_xor reduction.
// ---------------------------------------------------------------------------
__global__ __launch_bounds__(256) void l2norm_kernel(
    const float* __restrict__ in, const float* __restrict__ alpha,
    u16* __restrict__ outB, int rows)
{
    int wid = threadIdx.x >> 5, lane = threadIdx.x & 31;
    int row = blockIdx.x * 8 + wid;
    if (row >= rows) return;
    const float* rp = in + (long)row * R_REP;
    float a = alpha ? alpha[0] : 0.f;
    float x[8]; float s = 0.f;
    #pragma unroll
    for (int j = 0; j < 8; ++j) {
        float v = rp[lane + j * 32];
        if (alpha) v = (v >= 0.f) ? v : a * v;
        x[j] = v; s += v * v;
    }
    #pragma unroll
    for (int o = 16; o > 0; o >>= 1) s += __shfl_xor(s, o, 32);
    float inv = 1.f / fmaxf(sqrtf(s), 1e-12f);
    u16* op = outB + (long)row * R_REP;
    #pragma unroll
    for (int j = 0; j < 8; ++j) op[lane + j * 32] = f2bf(x[j] * inv);
}

// ---------------------------------------------------------------------------
// Gather + transpose: out[r*TOPK + j] = teach[sample[j]*R + r]  (bf16)
// ---------------------------------------------------------------------------
__global__ void gatherT_kernel(const u16* __restrict__ t,
                               const int* __restrict__ sample,
                               u16* __restrict__ out)
{
    int idx = blockIdx.x * blockDim.x + threadIdx.x;
    if (idx >= R_REP * TOPK) return;
    int r = idx / TOPK, j = idx % TOPK;
    out[idx] = t[(long)sample[j] * R_REP + r];
}

// ---------------------------------------------------------------------------
// Diffusion sims: for out-row i, n = not_zero[i]:
//   outq[i,k] = dot(teach[diff[n,k]], pred[n]) * scale
//   outk[i,k] = dot(teach[diff[n,k]], teach[n]) * scale
// One wave per row; shares the neighbor-row gather between q and k.
// ---------------------------------------------------------------------------
__global__ __launch_bounds__(256) void diff_kernel(
    const int* __restrict__ notz, const int* __restrict__ diff,
    const u16* __restrict__ predb, const u16* __restrict__ teachb,
    float* __restrict__ outq, float* __restrict__ outk, float scale)
{
    int wid = threadIdx.x >> 5, lane = threadIdx.x & 31;
    int i = blockIdx.x * 8 + wid;
    if (i >= N_NODES) return;
    int n = notz[i];
    float pv[8], tv[8];
    {
        H8 t;
        t.q = *(const uint4*)(predb + (long)n * R_REP + lane * 8);
        #pragma unroll
        for (int j = 0; j < 8; ++j) pv[j] = bf2f(t.u[j]);
        t.q = *(const uint4*)(teachb + (long)n * R_REP + lane * 8);
        #pragma unroll
        for (int j = 0; j < 8; ++j) tv[j] = bf2f(t.u[j]);
    }
    for (int k = 0; k < K_DIFF; ++k) {
        int d = diff[(long)n * K_DIFF + k];
        H8 t; t.q = *(const uint4*)(teachb + (long)d * R_REP + lane * 8);
        float sq = 0.f, sk = 0.f;
        #pragma unroll
        for (int j = 0; j < 8; ++j) {
            float dv = bf2f(t.u[j]);
            sq += dv * pv[j]; sk += dv * tv[j];
        }
        #pragma unroll
        for (int o = 16; o > 0; o >>= 1) {
            sq += __shfl_xor(sq, o, 32);
            sk += __shfl_xor(sk, o, 32);
        }
        if (lane == 0) {
            outq[(long)i * K_DIFF + k] = sq * scale;
            outk[(long)i * K_DIFF + k] = sk * scale;
        }
    }
}

// ---------------------------------------------------------------------------
extern "C" void kernel_launch(void* const* d_in, const int* in_sizes, int n_in,
                              void* d_out, int out_size, void* d_ws, size_t ws_size,
                              hipStream_t stream)
{
    (void)in_sizes; (void)n_in; (void)out_size; (void)ws_size;
    const float* x1  = (const float*)d_in[0];
    const float* x2  = (const float*)d_in[1];
    const int*   ei1 = (const int*)d_in[2];
    const int*   ei2 = (const int*)d_in[3];
    const float* ew1 = (const float*)d_in[4];
    const float* ew2 = (const float*)d_in[5];
    const int*   dif = (const int*)d_in[6];
    const int*   smp = (const int*)d_in[7];
    const int*   nz  = (const int*)d_in[8];
    const float* Ws1 = (const float*)d_in[9];
    const float* as1 = (const float*)d_in[10];
    const float* Ws2 = (const float*)d_in[11];
    const float* as2 = (const float*)d_in[12];
    const float* Wp1 = (const float*)d_in[13];
    const float* bp1 = (const float*)d_in[14];
    const float* ap  = (const float*)d_in[15];
    const float* Wp2 = (const float*)d_in[16];
    const float* bp2 = (const float*)d_in[17];
    const float* Wt1 = (const float*)d_in[18];
    const float* at1 = (const float*)d_in[19];
    const float* Wt2 = (const float*)d_in[20];
    const float* at2 = (const float*)d_in[21];

    float* out = (float*)d_out;
    const long NT = (long)N_NODES * TOPK, NK = (long)N_NODES * K_DIFF;
    float* o_q1  = out;
    float* o_q1d = o_q1 + NT;
    float* o_k1  = o_q1d + NK;
    float* o_k1d = o_k1 + NT;
    float* o_q2  = o_k1d + NK;
    float* o_q2d = o_q2 + NT;
    float* o_k2  = o_q2d + NK;
    float* o_k2d = o_k2 + NT;
    float* o_st1 = o_k2d + NK;      // student1 (N x R)

    // ---- workspace carve (256B aligned) ----
    char* p = (char*)d_ws;
    auto carve = [&](size_t bytes) { void* r = (void*)p; p += (bytes + 255) & ~(size_t)255; return r; };
    u16*   Xb   = (u16*)  carve((size_t)N_NODES * D_IN * 2);  // x bf16; reused as student bf16
    u16*   Hb   = (u16*)  carve((size_t)N_NODES * H_HID * 2); // hidden bf16
    float* lin  = (float*)carve((size_t)N_NODES * H_HID * 4); // GEMM f32 out
    float* agg  = (float*)carve((size_t)N_NODES * H_HID * 4); // segment-sum accumulator
    u16* Ws1b = (u16*)carve((size_t)D_IN * H_HID * 2);
    u16* Ws2b = (u16*)carve((size_t)H_HID * R_REP * 2);
    u16* Wp1b = (u16*)carve((size_t)R_REP * PH_DIM * 2);
    u16* Wp2b = (u16*)carve((size_t)PH_DIM * R_REP * 2);
    u16* Wt1b = (u16*)carve((size_t)D_IN * H_HID * 2);
    u16* Wt2b = (u16*)carve((size_t)H_HID * R_REP * 2);
    u16* t1b  = (u16*)carve((size_t)N_NODES * R_REP * 2);
    u16* t2b  = (u16*)carve((size_t)N_NODES * R_REP * 2);
    u16* p1b  = (u16*)carve((size_t)N_NODES * R_REP * 2);
    u16* p2b  = (u16*)carve((size_t)N_NODES * R_REP * 2);
    u16* t1sb = (u16*)carve((size_t)R_REP * TOPK * 2);
    u16* t2sb = (u16*)carve((size_t)R_REP * TOPK * 2);

    auto gemm = [&](const u16* A, const u16* B, int M, int K, int No,
                    const float* bias, const float* alpha, float scale,
                    float* oF, u16* oB) {
        dim3 g((M + 127) / 128, No / 64);
        gemm_bf16_wmma<<<g, 256, 0, stream>>>(A, B, bias, alpha, scale, oF, oB, M, K, No);
    };
    auto ew = [&](const float* in, const float* alpha, float* oF, u16* oB, long n) {
        ew_kernel<<<(unsigned)((n + 255) / 256), 256, 0, stream>>>(in, alpha, oF, oB, n);
    };
    auto eagg = [&](const int* ei, const float* w, const float* h, float* o, int F) {
        hipMemsetAsync(o, 0, (size_t)N_NODES * F * 4, stream);
        long tot = (long)E_EDGES * (F >> 2);
        edge_agg_kernel<<<(unsigned)((tot + 255) / 256), 256, 0, stream>>>(ei, w, h, o, E_EDGES, F);
    };
    const unsigned rowsGrid = (N_NODES + 7) / 8;

    // ---- weights -> bf16 ----
    ew(Ws1, nullptr, nullptr, Ws1b, (long)D_IN * H_HID);
    ew(Ws2, nullptr, nullptr, Ws2b, (long)H_HID * R_REP);
    ew(Wp1, nullptr, nullptr, Wp1b, (long)R_REP * PH_DIM);
    ew(Wp2, nullptr, nullptr, Wp2b, (long)PH_DIM * R_REP);
    ew(Wt1, nullptr, nullptr, Wt1b, (long)D_IN * H_HID);
    ew(Wt2, nullptr, nullptr, Wt2b, (long)H_HID * R_REP);

    for (int v = 0; v < 2; ++v) {
        const float* x   = v ? x2 : x1;
        const int*   ei  = v ? ei2 : ei1;
        const float* ewt = v ? ew2 : ew1;
        u16* teachb = v ? t2b : t1b;     // teacher1 pairs view1, teacher2 view2
        u16* predb  = v ? p2b : p1b;
        float* studF = v ? nullptr : o_st1;

        ew(x, nullptr, nullptr, Xb, (long)N_NODES * D_IN);

        // teacher encoder (runs first: frees Xb for reuse as student bf16)
        gemm(Xb, Wt1b, N_NODES, D_IN, H_HID, nullptr, nullptr, 1.f, lin, nullptr);
        eagg(ei, ewt, lin, agg, H_HID);
        ew(agg, at1, nullptr, Hb, (long)N_NODES * H_HID);
        gemm(Hb, Wt2b, N_NODES, H_HID, R_REP, nullptr, nullptr, 1.f, lin, nullptr);
        eagg(ei, ewt, lin, agg, R_REP);
        l2norm_kernel<<<rowsGrid, 256, 0, stream>>>(agg, at2, teachb, N_NODES);

        // student encoder
        gemm(Xb, Ws1b, N_NODES, D_IN, H_HID, nullptr, nullptr, 1.f, lin, nullptr);
        eagg(ei, ewt, lin, agg, H_HID);
        ew(agg, as1, nullptr, Hb, (long)N_NODES * H_HID);
        gemm(Hb, Ws2b, N_NODES, H_HID, R_REP, nullptr, nullptr, 1.f, lin, nullptr);
        eagg(ei, ewt, lin, agg, R_REP);
        u16* studb = Xb;                 // reuse Xb region for student bf16
        ew(agg, as2, studF, studb, (long)N_NODES * R_REP);

        // predictor: Linear(R,PH)+PReLU (fused) -> Linear(PH,R)+bias -> l2norm
        gemm(studb, Wp1b, N_NODES, R_REP, PH_DIM, bp1, ap, 1.f, nullptr, Hb);
        gemm(Hb, Wp2b, N_NODES, PH_DIM, R_REP, bp2, nullptr, 1.f, lin, nullptr);
        l2norm_kernel<<<rowsGrid, 256, 0, stream>>>(lin, nullptr, predb, N_NODES);
    }

    // ---- sampled-negative sims: A(Nx256) @ Bt(256x256), scale = 1/T = 2 ----
    gatherT_kernel<<<(R_REP * TOPK + 255) / 256, 256, 0, stream>>>(t2b, smp, t2sb);
    gatherT_kernel<<<(R_REP * TOPK + 255) / 256, 256, 0, stream>>>(t1b, smp, t1sb);
    gemm(p1b, t2sb, N_NODES, R_REP, TOPK, nullptr, nullptr, 2.f, o_q1, nullptr);
    gemm(t2b, t2sb, N_NODES, R_REP, TOPK, nullptr, nullptr, 2.f, o_k1, nullptr);
    gemm(p2b, t1sb, N_NODES, R_REP, TOPK, nullptr, nullptr, 2.f, o_q2, nullptr);
    gemm(t1b, t1sb, N_NODES, R_REP, TOPK, nullptr, nullptr, 2.f, o_k2, nullptr);

    // ---- diffusion sims (scale = 1/0.5 = 2) ----
    diff_kernel<<<rowsGrid, 256, 0, stream>>>(nz, dif, p1b, t2b, o_q1d, o_k1d, 2.f);
    diff_kernel<<<rowsGrid, 256, 0, stream>>>(nz, dif, p2b, t1b, o_q2d, o_k2d, 2.f);
}
